// base_model_20057497272912
// MI455X (gfx1250) — matrix operand order repaired and checked by reading, not compiled, not run on hardware
//
#include <hip/hip_runtime.h>
#include <hip/hip_bf16.h>

#define B_   32
#define T_   24
#define N_   512
#define IN_  64
#define H_   128
#define BN_  (B_ * N_)      // 16384
#define G3H  (3 * H_)       // 384
#define MT   4              // M-tiles (of 16 rows) per block

typedef __attribute__((ext_vector_type(16))) __bf16 v16bf;
typedef __attribute__((ext_vector_type(8)))  __bf16 v8bf;
typedef __attribute__((ext_vector_type(8)))  float  v8f;
typedef __attribute__((ext_vector_type(4)))  float  v4f;

union V16BF { v16bf v; v8bf h[2]; };

static __device__ inline v8f zero8() {
    v8f z;
#pragma unroll
    for (int i = 0; i < 8; ++i) z[i] = 0.0f;
    return z;
}

// A-matrix fragment (16x32 bf16): lane holds row M = lane%16,
// K = kb..kb+7 and kb+16..kb+23 where kb = f*32 + (lane/16)*8.
static __device__ inline v16bf loadA_bf(const __bf16* base, int stride, int row, int kb) {
    V16BF u;
    const __bf16* p = base + (size_t)row * stride + kb;
    u.h[0] = *(const v8bf*)(p);
    u.h[1] = *(const v8bf*)(p + 16);
    return u.v;
}

// B-matrix fragment (32x16 bf16), B[k][n] = W[n][k] for row-major W:
// lane holds column n = lane%16, K = kbB..kbB+7 and kbB+16..kbB+23, kbB = f*32 + (lane/16)*8.
// Two 8-bf16 = 16-byte loads, 16B-aligned here.
static __device__ inline v16bf loadB_bf(const __bf16* base, int stride, int row, int kbB) {
    V16BF u;
    const __bf16* p = base + (size_t)row * stride + kbB;
    u.h[0] = *(const v8bf*)(p);
    u.h[1] = *(const v8bf*)(p + 16);
    return u.v;
}

static __device__ inline v8f wmma_bf16(v16bf a, v16bf b, v8f c) {
    return __builtin_amdgcn_wmma_f32_16x16x32_bf16(
        /*neg_a=*/false, a, /*neg_b=*/false, b,
        /*c_mod=*/(short)0, c, /*reuse_a=*/false, /*reuse_b=*/false);
}

static __device__ inline float fast_sigmoid(float x) {
    return 1.0f / (1.0f + __expf(-x));
}
static __device__ inline float fast_tanh(float x) {
    return 2.0f / (1.0f + __expf(-2.0f * x)) - 1.0f;
}

// ---------------------------------------------------------------------------
// Prep: fp32 weights -> bf16 copies in workspace; zero h0 (f32 + bf16).
// ---------------------------------------------------------------------------
__global__ void prep_kernel(const float* __restrict__ W_mlp,
                            const float* __restrict__ W_ih,
                            const float* __restrict__ W_hh,
                            __bf16* __restrict__ w_mlp_bf,
                            __bf16* __restrict__ w_ih_bf,
                            __bf16* __restrict__ w_hh_bf,
                            float* __restrict__ h0_f32,
                            __bf16* __restrict__ h0_bf) {
    int idx = blockIdx.x * blockDim.x + threadIdx.x;
    int stride = gridDim.x * blockDim.x;
    for (int i = idx; i < H_ * IN_; i += stride) w_mlp_bf[i] = (__bf16)W_mlp[i];
    for (int i = idx; i < G3H * H_; i += stride) w_ih_bf[i] = (__bf16)W_ih[i];
    for (int i = idx; i < G3H * H_; i += stride) w_hh_bf[i] = (__bf16)W_hh[i];
    for (int i = idx; i < BN_ * H_; i += stride) {
        h0_f32[i] = 0.0f;
        h0_bf[i]  = (__bf16)0.0f;
    }
}

// ---------------------------------------------------------------------------
// One GRU step, MLP fused. Block = 256 threads = 8 waves, handles MT
// 16-row M-tiles (64 rows). Wave w owns hidden-column tile [w*16, w*16+16).
// Weight B-fragments are loaded once per K-step and reused across all MT
// M-tiles (4x reduction of weight read traffic vs one tile per block).
// ---------------------------------------------------------------------------
__global__ __launch_bounds__(256)
void gru_step_kernel(const float* __restrict__ x, int t,
                     const __bf16* __restrict__ w_mlp_bf,
                     const float* __restrict__ b_mlp,
                     const __bf16* __restrict__ w_ih_bf,
                     const __bf16* __restrict__ w_hh_bf,
                     const float* __restrict__ b_ih,
                     const float* __restrict__ b_hh,
                     const float* __restrict__ h_in_f32,
                     const __bf16* __restrict__ h_in_bf,
                     float* __restrict__ h_out_f32,
                     __bf16* __restrict__ h_out_bf) {
    const int LSTR = H_ + 8;  // 136 bf16 = 272 B row stride (16B aligned, bank-staggered)
    __shared__ __attribute__((aligned(16))) __bf16 lds_emb[16 * MT * (H_ + 8)];

    const int lane    = threadIdx.x & 31;
    const int w       = threadIdx.x >> 5;      // wave id == column tile id, 0..7
    const int halfsel = lane >> 4;             // 0: lanes 0-15, 1: lanes 16-31
    const int lcol    = lane & 15;
    const int m0      = blockIdx.x * (16 * MT);  // rows m0..m0+63 of the BN batch

    // ---- Stage 1: emb = relu(x_slab @ W_mlp^T + b_mlp) for MT M-tiles ----
    // 64 | N_ = 512 => all 64 rows share b and have consecutive n.
    const int b  = m0 / N_;
    const int n0 = m0 % N_;
    const float* xbase = x + (((size_t)b * T_ + t) * N_ + n0) * IN_;

    // W_mlp B-fragments: loaded once, reused for all MT tiles.
    v16bf bw[2];
#pragma unroll
    for (int f = 0; f < 2; ++f)
        bw[f] = loadB_bf(w_mlp_bf, IN_, w * 16 + lcol, f * 32 + halfsel * 8);
    const float bias_e = b_mlp[w * 16 + lcol];

#pragma unroll
    for (int mt = 0; mt < MT; ++mt) {
        v8f acc_e = zero8();
#pragma unroll
        for (int f = 0; f < 2; ++f) {
            const int kbA = f * 32 + halfsel * 8;
            v16bf a;
            {
                const float* p = xbase + (size_t)(mt * 16 + lcol) * IN_ + kbA;
                v4f q0 = *(const v4f*)(p);
                v4f q1 = *(const v4f*)(p + 4);
                v4f q2 = *(const v4f*)(p + 16);
                v4f q3 = *(const v4f*)(p + 20);
#pragma unroll
                for (int i = 0; i < 4; ++i) {
                    a[i]      = (__bf16)q0[i];
                    a[4 + i]  = (__bf16)q1[i];
                    a[8 + i]  = (__bf16)q2[i];
                    a[12 + i] = (__bf16)q3[i];
                }
            }
            acc_e = wmma_bf16(a, bw[f], acc_e);
        }
#pragma unroll
        for (int v = 0; v < 8; ++v) {
            float e = acc_e[v] + bias_e;
            e = e > 0.0f ? e : 0.0f;
            const int row = mt * 16 + v + halfsel * 8;
            lds_emb[row * LSTR + w * 16 + lcol] = (__bf16)e;
        }
    }
    __syncthreads();

    // ---- Stage 2: gates. Accumulators per M-tile; B-frags loaded once per
    //      K-step and reused across the MT tiles. ----
    v8f acc_r[MT], acc_z[MT], acc_in[MT], acc_hn[MT];
#pragma unroll
    for (int mt = 0; mt < MT; ++mt) {
        acc_r[mt]  = zero8();
        acc_z[mt]  = zero8();
        acc_in[mt] = zero8();
        acc_hn[mt] = zero8();
    }

    const int cr = 0 * H_ + w * 16 + lcol;
    const int cz = 1 * H_ + w * 16 + lcol;
    const int cn = 2 * H_ + w * 16 + lcol;

#pragma unroll
    for (int f = 0; f < 4; ++f) {
        const int kbA = f * 32 + halfsel * 8;
        const int kbB = f * 32 + halfsel * 8;

        const v16bf b_ir = loadB_bf(w_ih_bf, H_, cr, kbB);
        const v16bf b_hr = loadB_bf(w_hh_bf, H_, cr, kbB);
        const v16bf b_iz = loadB_bf(w_ih_bf, H_, cz, kbB);
        const v16bf b_hz = loadB_bf(w_hh_bf, H_, cz, kbB);
        const v16bf b_in = loadB_bf(w_ih_bf, H_, cn, kbB);
        const v16bf b_hn = loadB_bf(w_hh_bf, H_, cn, kbB);

#pragma unroll
        for (int mt = 0; mt < MT; ++mt) {
            // A_emb from LDS (row M = lane%16 within tile mt).
            v16bf a_e;
            {
                V16BF u;
                const __bf16* p = &lds_emb[(mt * 16 + lcol) * LSTR + kbA];
                u.h[0] = *(const v8bf*)(p);
                u.h[1] = *(const v8bf*)(p + 16);
                a_e = u.v;
            }
            // A_h from global bf16 hidden state.
            const v16bf a_h = loadA_bf(h_in_bf, H_, m0 + mt * 16 + lcol, kbA);

            acc_r[mt]  = wmma_bf16(a_e, b_ir, acc_r[mt]);
            acc_r[mt]  = wmma_bf16(a_h, b_hr, acc_r[mt]);
            acc_z[mt]  = wmma_bf16(a_e, b_iz, acc_z[mt]);
            acc_z[mt]  = wmma_bf16(a_h, b_hz, acc_z[mt]);
            acc_in[mt] = wmma_bf16(a_e, b_in, acc_in[mt]);
            acc_hn[mt] = wmma_bf16(a_h, b_hn, acc_hn[mt]);
        }
    }

    // ---- Gate math + h update (fp32) ----
    const int c0 = w * 16 + lcol;
    const float br    = b_ih[c0]            + b_hh[c0];
    const float bz    = b_ih[H_ + c0]       + b_hh[H_ + c0];
    const float bin_i = b_ih[2 * H_ + c0];
    const float bin_h = b_hh[2 * H_ + c0];

#pragma unroll
    for (int mt = 0; mt < MT; ++mt) {
#pragma unroll
        for (int v = 0; v < 8; ++v) {
            const int row = m0 + mt * 16 + v + halfsel * 8;
            const float r  = fast_sigmoid(acc_r[mt][v] + br);
            const float z  = fast_sigmoid(acc_z[mt][v] + bz);
            const float n  = fast_tanh(acc_in[mt][v] + bin_i + r * (acc_hn[mt][v] + bin_h));
            const float hp = h_in_f32[(size_t)row * H_ + c0];
            const float hn = (1.0f - z) * n + z * hp;
            h_out_f32[(size_t)row * H_ + c0] = hn;
            h_out_bf [(size_t)row * H_ + c0] = (__bf16)hn;
        }
    }
}

// ---------------------------------------------------------------------------
extern "C" void kernel_launch(void* const* d_in, const int* in_sizes, int n_in,
                              void* d_out, int out_size, void* d_ws, size_t ws_size,
                              hipStream_t stream) {
    (void)in_sizes; (void)n_in; (void)out_size; (void)ws_size;

    const float* x     = (const float*)d_in[0];
    const float* W_mlp = (const float*)d_in[1];
    const float* b_mlp = (const float*)d_in[2];
    const float* W_ih  = (const float*)d_in[3];
    const float* W_hh  = (const float*)d_in[4];
    const float* b_ih  = (const float*)d_in[5];
    const float* b_hh  = (const float*)d_in[6];

    char* ws = (char*)d_ws;
    size_t off = 0;
    auto alloc = [&](size_t bytes) -> void* {
        void* p = ws + off;
        off = (off + bytes + 255) & ~((size_t)255);
        return p;
    };

    __bf16* w_mlp_bf = (__bf16*)alloc((size_t)H_ * IN_ * 2);
    __bf16* w_ih_bf  = (__bf16*)alloc((size_t)G3H * H_ * 2);
    __bf16* w_hh_bf  = (__bf16*)alloc((size_t)G3H * H_ * 2);
    __bf16* h_bf0    = (__bf16*)alloc((size_t)BN_ * H_ * 2);
    __bf16* h_bf1    = (__bf16*)alloc((size_t)BN_ * H_ * 2);
    float*  h_f0     = (float*) alloc((size_t)BN_ * H_ * 4);
    float*  h_f1     = (float*) alloc((size_t)BN_ * H_ * 4);

    __bf16* h_bf[2]  = { h_bf0, h_bf1 };
    float*  h_f32[2] = { h_f0,  h_f1 };

    prep_kernel<<<1024, 256, 0, stream>>>(W_mlp, W_ih, W_hh,
                                          w_mlp_bf, w_ih_bf, w_hh_bf,
                                          h_f32[0], h_bf[0]);

    for (int t = 0; t < T_; ++t) {
        const int cur = t & 1;
        const int nxt = cur ^ 1;
        float* out_f32 = (t == T_ - 1) ? (float*)d_out : h_f32[nxt];
        gru_step_kernel<<<BN_ / (16 * MT), 256, 0, stream>>>(
            x, t, w_mlp_bf, b_mlp, w_ih_bf, w_hh_bf, b_ih, b_hh,
            h_f32[cur], h_bf[cur], out_f32, h_bf[nxt]);
    }
}
